// Attention_68092411511590
// MI455X (gfx1250) — matrix operand order repaired and checked by reading, not compiled, not run on hardware
//
#include <hip/hip_runtime.h>
#include <hip/hip_bf16.h>
#include <float.h>

#define HEADS 8

typedef __attribute__((ext_vector_type(16))) __bf16        v16bf;
typedef __attribute__((ext_vector_type(8)))  float         v8f;
typedef __attribute__((ext_vector_type(4)))  unsigned int  u32x4;
typedef __attribute__((ext_vector_type(2)))  unsigned int  u32x2;
typedef __attribute__((ext_vector_type(4)))  float         f32x4;

union FragAB { u32x4 u[2]; v16bf v; };   // 32 bytes = 16 bf16 = one WMMA A/B operand

__device__ __forceinline__ unsigned short f2bf(float f) {
    union { float f; unsigned int u; } c; c.f = f;
    unsigned int u = c.u + 0x7FFFu + ((c.u >> 16) & 1u);   // round-to-nearest-even
    return (unsigned short)(u >> 16);
}

// ---------------------------------------------------------------------------
// Pack fp32 weight [K][N] (row major) into bf16 B-fragment layout:
//   Bp[kt][n][j] = W[kt*32 + j][n]   (kt = K/32 tiles, j = K-within-tile)
// A lane's B fragment (col n, K 0..15 or 16..31) is then 32 contiguous bytes.
// ---------------------------------------------------------------------------
__global__ void pack_w_kernel(const float* __restrict__ W,
                              unsigned short* __restrict__ Bp,
                              int K, int N) {
    int tid = blockIdx.x * blockDim.x + threadIdx.x;
    if (tid >= K * N) return;
    int j    = tid & 31;
    int rest = tid >> 5;
    int n    = rest % N;
    int kt   = rest / N;
    Bp[tid] = f2bf(W[(kt * 32 + j) * N + n]);
}

// ---------------------------------------------------------------------------
// Fully fused attention: one workgroup (8 waves) per (b, s) tile of 16 frames.
// LDS map (80 KB dynamic):
//   AO  : bf16[16][512]   x-tile (phase 1-2), reused as attn-out (phase 3-4)
//   QK  : bf16[16][1024]  cols 0..511 = q*scale, 512..1023 = k
//   VT  : bf16[8][64][16] v transposed per head: VT[h][d][frame]
//   Sb  : f32 [8][16][16] raw scores
//   Pb  : bf16[8][16][32] softmax probs, zero-padded to K=32
// Each wave processes TWO N-tiles per k-loop: A fragment loaded once from LDS
// feeds two independent WMMA accumulator chains (better latency hiding,
// half the DS traffic, and back-to-back WMMAs sharing the A operand).
// ---------------------------------------------------------------------------
__global__ void fused_attn_kernel(const float* __restrict__ x,
                                  const float* __restrict__ pos_bias,
                                  const int*   __restrict__ fmask,
                                  const unsigned short* __restrict__ wqkv_p,
                                  const unsigned short* __restrict__ wout_p,
                                  float* __restrict__ out) {
    extern __shared__ char smem_raw[];
    unsigned short* AO = (unsigned short*)smem_raw;       // 8192 ushorts
    unsigned short* QK = AO + 8192;                       // 16384 ushorts
    unsigned short* VT = AO + 24576;                      // 8192 ushorts
    float*          Sb = (float*)(smem_raw + 65536);      // 2048 floats
    unsigned short* Pb = AO + 36864;                      // 4096 ushorts

    const int tid  = threadIdx.x;
    const int wave = tid >> 5;
    const int lane = tid & 31;
    const int m    = lane & 15;        // A-row / B-col / C-col index
    const int hi   = lane >> 4;        // lane half
    const int aoff = hi * 16;          // A-fragment K sub-offset

    const int bb = blockIdx.x >> 10;   // batch
    const float* xt = x + (size_t)blockIdx.x * (16 * 512);

    // ---- Phase 1: x tile fp32 -> bf16 in LDS ------------------------------
    for (int i = tid; i < 2048; i += 256) {
        f32x4 v4 = ((const f32x4*)xt)[i];
        u32x2 pk;
        pk.x = (unsigned int)f2bf(v4.x) | ((unsigned int)f2bf(v4.y) << 16);
        pk.y = (unsigned int)f2bf(v4.z) | ((unsigned int)f2bf(v4.w) << 16);
        *(u32x2*)&AO[i * 4] = pk;
    }
    __syncthreads();

    // ---- Phase 2: QKV GEMM  [16,512] @ [512,1536] -------------------------
    // 96 N-tiles / 8 waves = 12 per wave, processed as 6 pairs.
    for (int u = 0; u < 6; ++u) {
        const int nb0 = (wave + u * 16) << 4;     // tile pair: nb0, nb0+128
        v8f acc0 = {}, acc1 = {};
#pragma unroll
        for (int kt = 0; kt < 16; ++kt) {
            FragAB A, B0, B1;
            const int kb = kt << 5;
            A.u[0] = *(const u32x4*)&AO[m * 512 + kb + aoff];
            A.u[1] = *(const u32x4*)&AO[m * 512 + kb + 8 + aoff];
            const unsigned short* bp =
                wqkv_p + ((size_t)(kt * 1536 + nb0 + m)) * 32 + (hi ? 16 : 0);
            B0.u[0] = *(const u32x4*)(bp);
            B0.u[1] = *(const u32x4*)(bp + 8);
            B1.u[0] = *(const u32x4*)(bp + 128 * 32);
            B1.u[1] = *(const u32x4*)(bp + 128 * 32 + 8);
            acc0 = __builtin_amdgcn_wmma_f32_16x16x32_bf16(
                false, A.v, false, B0.v, (short)0, acc0, false, false);
            acc1 = __builtin_amdgcn_wmma_f32_16x16x32_bf16(
                false, A.v, false, B1.v, (short)0, acc1, false, false);
        }
#pragma unroll
        for (int p = 0; p < 2; ++p) {
            const int col = nb0 + p * 128 + m;
            const v8f acc = p ? acc1 : acc0;
#pragma unroll
            for (int r = 0; r < 8; ++r) {
                const int row = r + hi * 8;
                const float val = acc[r];
                if (col < 512) {                        // q (pre-scaled 64^-0.5)
                    QK[row * 1024 + col] = f2bf(val * 0.125f);
                } else if (col < 1024) {                // k
                    QK[row * 1024 + col] = f2bf(val);
                } else {                                // v, stored transposed
                    const int d = col - 1024;
                    VT[((d >> 6) * 64 + (d & 63)) * 16 + row] = f2bf(val);
                }
            }
        }
    }
    __syncthreads();

    // ---- Phase 3: per-head attention (wave == head) -----------------------
    const int h = wave;
    {
        v8f s = {};
#pragma unroll
        for (int kt = 0; kt < 2; ++kt) {
            FragAB A, B;
            const int qc = h * 64 + kt * 32;
            A.u[0] = *(const u32x4*)&QK[m * 1024 + qc + aoff];
            A.u[1] = *(const u32x4*)&QK[m * 1024 + qc + 8 + aoff];
            const int kc = 512 + h * 64 + kt * 32 + hi * 16;   // k row = B col = m
            B.u[0] = *(const u32x4*)&QK[m * 1024 + kc];
            B.u[1] = *(const u32x4*)&QK[m * 1024 + kc + 8];
            s = __builtin_amdgcn_wmma_f32_16x16x32_bf16(
                false, A.v, false, B.v, (short)0, s, false, false);
        }
#pragma unroll
        for (int r = 0; r < 8; ++r)
            Sb[h * 256 + (r + hi * 8) * 16 + m] = s[r];
    }

    // softmax: one row per lane (lanes 0..15), + pos_bias + focus mask
    if (lane < 16) {
        const int i = lane;
        const bool focus = fmask[bb] != 0;
        float vals[16];
        float mx = -FLT_MAX;
#pragma unroll
        for (int j = 0; j < 16; ++j) {
            float sv = Sb[h * 256 + i * 16 + j] + pos_bias[h * 256 + i * 16 + j];
            if (focus && (i != j)) sv = -FLT_MAX;
            vals[j] = sv;
            mx = fmaxf(mx, sv);
        }
        float sum = 0.f;
#pragma unroll
        for (int j = 0; j < 16; ++j) { vals[j] = __expf(vals[j] - mx); sum += vals[j]; }
        const float inv = 1.0f / sum;
#pragma unroll
        for (int j = 0; j < 16; ++j) Pb[h * 512 + i * 32 + j] = f2bf(vals[j] * inv);
#pragma unroll
        for (int j = 16; j < 32; ++j) Pb[h * 512 + i * 32 + j] = 0;  // pad K 16..31
    }

    // P @ V : K padded to 32 (upper half zero); 4 column tiles of 16
    {
        FragAB A;
        A.u[0] = *(const u32x4*)&Pb[h * 512 + m * 32 + aoff];
        A.u[1] = *(const u32x4*)&Pb[h * 512 + m * 32 + 8 + aoff];
#pragma unroll
        for (int nt = 0; nt < 4; ++nt) {
            FragAB B;
            u32x4 b0 = 0, b1 = 0;
            if (hi == 0) {                        // lanes 16..31: K=16..31 are pad
                const int d = nt * 16 + m;
                b0 = *(const u32x4*)&VT[(h * 64 + d) * 16];
                b1 = *(const u32x4*)&VT[(h * 64 + d) * 16 + 8];
            }
            B.u[0] = b0; B.u[1] = b1;
            v8f o = {};
            o = __builtin_amdgcn_wmma_f32_16x16x32_bf16(
                false, A.v, false, B.v, (short)0, o, false, false);
#pragma unroll
            for (int r = 0; r < 8; ++r)           // AO reused as attn output
                AO[(r + hi * 8) * 512 + h * 64 + nt * 16 + m] = f2bf(o[r]);
        }
    }
    __syncthreads();

    // ---- Phase 4: output GEMM  [16,512] @ [512,512] -> global -------------
    // 32 N-tiles / 8 waves = 4 per wave, processed as 2 pairs.
    float* outp = out + (size_t)blockIdx.x * (16 * 512);
    for (int u = 0; u < 2; ++u) {
        const int nb0 = (wave + u * 16) << 4;     // tile pair: nb0, nb0+128
        v8f acc0 = {}, acc1 = {};
#pragma unroll
        for (int kt = 0; kt < 16; ++kt) {
            FragAB A, B0, B1;
            const int kb = kt << 5;
            A.u[0] = *(const u32x4*)&AO[m * 512 + kb + aoff];
            A.u[1] = *(const u32x4*)&AO[m * 512 + kb + 8 + aoff];
            const unsigned short* bp =
                wout_p + ((size_t)(kt * 512 + nb0 + m)) * 32 + (hi ? 16 : 0);
            B0.u[0] = *(const u32x4*)(bp);
            B0.u[1] = *(const u32x4*)(bp + 8);
            B1.u[0] = *(const u32x4*)(bp + 128 * 32);
            B1.u[1] = *(const u32x4*)(bp + 128 * 32 + 8);
            acc0 = __builtin_amdgcn_wmma_f32_16x16x32_bf16(
                false, A.v, false, B0.v, (short)0, acc0, false, false);
            acc1 = __builtin_amdgcn_wmma_f32_16x16x32_bf16(
                false, A.v, false, B1.v, (short)0, acc1, false, false);
        }
#pragma unroll
        for (int p = 0; p < 2; ++p) {
            const int col = nb0 + p * 128 + m;
            const v8f acc = p ? acc1 : acc0;
#pragma unroll
            for (int r = 0; r < 8; ++r)
                outp[(r + hi * 8) * 512 + col] = acc[r];
        }
    }
}

// ---------------------------------------------------------------------------
extern "C" void kernel_launch(void* const* d_in, const int* in_sizes, int n_in,
                              void* d_out, int out_size, void* d_ws, size_t ws_size,
                              hipStream_t stream) {
    const float* x    = (const float*)d_in[0];   // [4,1024,16,512]
    const float* pb   = (const float*)d_in[1];   // [8,16,16]
    const int*   fm   = (const int*)d_in[2];     // [4]
    const float* wqkv = (const float*)d_in[3];   // [512,1536]
    const float* wout = (const float*)d_in[4];   // [512,512]
    float* out = (float*)d_out;                  // [4,1024,16,512]

    unsigned short* wqkv_p = (unsigned short*)d_ws;            // 512*1536 bf16
    unsigned short* wout_p = wqkv_p + 512 * 1536;              // 512*512  bf16

    pack_w_kernel<<<(512 * 1536 + 255) / 256, 256, 0, stream>>>(wqkv, wqkv_p, 512, 1536);
    pack_w_kernel<<<(512 * 512 + 255) / 256, 256, 0, stream>>>(wout, wout_p, 512, 512);

    fused_attn_kernel<<<4096, 256, 81920, stream>>>(x, pb, fm, wqkv_p, wout_p, out);
}
